// BahdanauAttention_5806795784198
// MI455X (gfx1250) — compile-verified, matrix operand order checked by reading, not run on hardware
//
#include <hip/hip_runtime.h>
#include <math.h>

// B=64, L=2048, D=512
constexpr int Bn = 64;
constexpr int Ln = 2048;
constexpr int Dn = 512;
constexpr int BSTRIDE = 520;   // 512 + 8 bf16 pad -> lanes land on distinct LDS bank groups

typedef __attribute__((ext_vector_type(16))) __bf16 v16bf;
typedef __attribute__((ext_vector_type(8)))  __bf16 v8bf;
typedef __attribute__((ext_vector_type(8)))  float  v8f;
typedef __attribute__((ext_vector_type(4)))  int    v4i;

#if defined(__has_builtin)
#  if __has_builtin(__builtin_amdgcn_global_load_async_to_lds_b128)
#    define USE_ASYNC_LDS 1
#  endif
#endif
#ifndef USE_ASYNC_LDS
#  define USE_ASYNC_LDS 0
#endif

__device__ inline void wait_async_zero() {
#if defined(__has_builtin) && __has_builtin(__builtin_amdgcn_s_wait_asynccnt)
  __builtin_amdgcn_s_wait_asynccnt(0);
#else
  asm volatile("s_wait_asynccnt 0x0" ::: "memory");
#endif
}

// ---------------- prep kernels (tiny) ----------------

// WhT_bf16[e*D + d] = bf16(Wh[d*D + e])  -> B fragments become contiguous-K loads
__global__ __launch_bounds__(256) void prep_wht(const float* __restrict__ Wh,
                                                __bf16* __restrict__ whT) {
  const int e = blockIdx.x;
  for (int d = threadIdx.x; d < Dn; d += 256)
    whT[(size_t)e * Dn + d] = (__bf16)Wh[(size_t)d * Dn + e];
}

// v[d] = sum_e Vw[d,e]   (folds the second GEMM + reduce_sum into one vector)
__global__ __launch_bounds__(256) void prep_v(const float* __restrict__ Vw,
                                              float* __restrict__ v) {
  const int d = blockIdx.x * 256 + threadIdx.x;
  if (d < Dn) {
    float s = 0.f;
    for (int e = 0; e < Dn; ++e) s += Vw[(size_t)d * Dn + e];
    v[d] = s;
  }
}

// dec[b,e] = s_t[b,:] @ Ws[:,e] + bs[e]
__global__ __launch_bounds__(256) void prep_dec(const float* __restrict__ s_t,
                                                const float* __restrict__ Ws,
                                                const float* __restrict__ bs,
                                                float* __restrict__ dec) {
  const int b = blockIdx.x;
  __shared__ float ss[Dn];
  for (int d = threadIdx.x; d < Dn; d += 256) ss[d] = s_t[(size_t)b * Dn + d];
  __syncthreads();
  for (int e = threadIdx.x; e < Dn; e += 256) {
    float acc = bs[e];
    for (int d = 0; d < Dn; ++d) acc += ss[d] * Ws[(size_t)d * Dn + e];
    dec[(size_t)b * Dn + e] = acc;
  }
}

// ---------------- fragment loaders (ISA 7.12.2 layouts) ----------------
// 16-bit A 16x32: lane m = lane&15; halves 0..7 -> K = kbase+0..7, halves 8..15 -> K = 16+kbase+0..7
// where kbase = (lane<16 ? 0 : 8). B (32x16 KxN) mirrored with N = lane&15.

__device__ inline v16bf load_afrag_f32(const float* __restrict__ p) {
  const float4* pa = (const float4*)p;          // K = kb+0..7
  const float4* pb = (const float4*)(p + 16);   // K = 16+kb+0..7
  float4 a0 = pa[0], a1 = pa[1], a2 = pb[0], a3 = pb[1];
  v16bf r;
  r[0]=(__bf16)a0.x; r[1]=(__bf16)a0.y; r[2]=(__bf16)a0.z; r[3]=(__bf16)a0.w;
  r[4]=(__bf16)a1.x; r[5]=(__bf16)a1.y; r[6]=(__bf16)a1.z; r[7]=(__bf16)a1.w;
  r[8]=(__bf16)a2.x; r[9]=(__bf16)a2.y; r[10]=(__bf16)a2.z; r[11]=(__bf16)a2.w;
  r[12]=(__bf16)a3.x; r[13]=(__bf16)a3.y; r[14]=(__bf16)a3.z; r[15]=(__bf16)a3.w;
  return r;
}

__device__ inline v16bf load_bfrag_lds(const __bf16* p) {
  v8bf lo = *(const v8bf*)p;          // K = kb+0..7   (ds_load_b128)
  v8bf hh = *(const v8bf*)(p + 16);   // K = 16+kb+0..7
  v16bf r;
#pragma unroll
  for (int i = 0; i < 8; ++i) { r[i] = lo[i]; r[i + 8] = hh[i]; }
  return r;
}

// ---------------- main fused kernel: enc GEMM + tanh + score dot ----------------
// grid = 1024 blocks (128 rows each), 256 threads = 8 waves, wave w -> 16 rows, full N=512.
// Per N-chunk (64 cols), the whole block stages the 64x512 bf16 B-slab in LDS once
// (async global->LDS when available), then all 8 waves consume it via ds_load_b128.
__global__ __launch_bounds__(256) void attn_scores(
    const float* __restrict__ h_i, const float* __restrict__ cov_in,
    const __bf16* __restrict__ whT, const float* __restrict__ dec,
    const float* __restrict__ Wc, const float* __restrict__ bh,
    const float* __restrict__ bc, const float* __restrict__ v,
    float* __restrict__ scores) {
  __shared__ __bf16 s_B[64 * BSTRIDE];              // 66,560 B
  __shared__ float s_dec[Dn], s_wc[Dn], s_bias[Dn], s_v[Dn], s_cov[128];

  const int tid = threadIdx.x;
  const int row_base = blockIdx.x * 128;   // flattened (b*L + l), block within one b
  const int b = row_base >> 11;            // /2048
  for (int i = tid; i < Dn; i += 256) {
    s_dec[i]  = dec[(size_t)b * Dn + i];
    s_wc[i]   = Wc[i];
    s_bias[i] = bh[i] + bc[i];
    s_v[i]    = v[i];
  }
  if (tid < 128) s_cov[tid] = cov_in[row_base + tid];

  const int wave = tid >> 5;
  const int lane = tid & 31;
  const int ml = lane & 15;
  const int hi = lane >> 4;
  const int kb = hi << 3;   // 0 or 8

  const float* Arow = h_i + (size_t)(row_base + wave * 16 + ml) * Dn;
  float sacc[8] = {0.f, 0.f, 0.f, 0.f, 0.f, 0.f, 0.f, 0.f};

  for (int nch = 0; nch < 8; ++nch) {          // 8 chunks of 64 output columns
    // ---- stage B slab: Wh^T rows [nch*64, nch*64+64), all K=512, into LDS ----
    const int e0 = nch * 64;
#pragma unroll
    for (int i = 0; i < 16; ++i) {             // 4096 x 16B chunks / 256 threads
      const int idx = tid + i * 256;
      const int row = idx >> 6;
      const int c8  = (idx & 63) << 3;
      const __bf16* src = whT + (size_t)(e0 + row) * Dn + c8;
      __bf16* dst = s_B + row * BSTRIDE + c8;
#if USE_ASYNC_LDS
      __builtin_amdgcn_global_load_async_to_lds_b128((v4i*)src, (v4i*)dst, 0, 0);
#else
      *(v8bf*)dst = *(const v8bf*)src;
#endif
    }
#if USE_ASYNC_LDS
    wait_async_zero();
#endif
    __syncthreads();

    v8f cacc[4] = {};
    const __bf16* Bcol = s_B + (size_t)ml * BSTRIDE;
#pragma unroll 4
    for (int kt = 0; kt < 16; ++kt) {          // K = 512 in steps of 32
      const int k0 = kt * 32;
      v16bf afrag = load_afrag_f32(Arow + k0 + kb);
      // hoist all 4 B-fragment LDS loads into distinct registers so the
      // ds_load_b128 clause overlaps the multi-cycle WMMAs (no dscnt-0 stalls)
      v16bf bfr[4];
#pragma unroll
      for (int t = 0; t < 4; ++t)
        bfr[t] = load_bfrag_lds(Bcol + (size_t)(t * 16) * BSTRIDE + k0 + kb);
#pragma unroll
      for (int t = 0; t < 4; ++t)
        cacc[t] = __builtin_amdgcn_wmma_f32_16x16x32_bf16(
            false, afrag, false, bfr[t], (short)0, cacc[t], false, false);
    }
    // fused epilogue: tanh(enc + dec + cov*Wc + (bh+bc)) * v  -> per-row partial score
    const int lr0 = wave * 16 + hi * 8;
#pragma unroll
    for (int t = 0; t < 4; ++t) {
      const int e = e0 + t * 16 + ml;
      const float de = s_dec[e], wc = s_wc[e], bi = s_bias[e], vv = s_v[e];
#pragma unroll
      for (int r = 0; r < 8; ++r) {
        const float val = tanhf(cacc[t][r] + de + s_cov[lr0 + r] * wc + bi);
        sacc[r] += val * vv;
      }
    }
    __syncthreads();   // protect s_B before next chunk restages
  }
  // reduce each row's partial score across the 16 lanes that hold its columns
#pragma unroll
  for (int r = 0; r < 8; ++r) {
    float x = sacc[r];
    x += __shfl_xor(x, 1, 16);
    x += __shfl_xor(x, 2, 16);
    x += __shfl_xor(x, 4, 16);
    x += __shfl_xor(x, 8, 16);
    if (ml == 0) scores[row_base + wave * 16 + hi * 8 + r] = x;
  }
}

// ---------------- softmax over L + coverage update (in-place on score region) ----------------
__global__ __launch_bounds__(256) void softmax_cov(const float* __restrict__ cov_in,
                                                   float* __restrict__ a_reg,
                                                   float* __restrict__ cov_out) {
  __shared__ float red[256];
  const int b = blockIdx.x, tid = threadIdx.x;
  float s[8];
#pragma unroll
  for (int j = 0; j < 8; ++j) s[j] = a_reg[(size_t)b * Ln + j * 256 + tid];
  float mx = s[0];
#pragma unroll
  for (int j = 1; j < 8; ++j) mx = fmaxf(mx, s[j]);
  red[tid] = mx; __syncthreads();
  for (int off = 128; off > 0; off >>= 1) {
    if (tid < off) red[tid] = fmaxf(red[tid], red[tid + off]);
    __syncthreads();
  }
  mx = red[0]; __syncthreads();
  float sum = 0.f;
#pragma unroll
  for (int j = 0; j < 8; ++j) { s[j] = __expf(s[j] - mx); sum += s[j]; }
  red[tid] = sum; __syncthreads();
  for (int off = 128; off > 0; off >>= 1) {
    if (tid < off) red[tid] += red[tid + off];
    __syncthreads();
  }
  const float inv = 1.0f / red[0];
#pragma unroll
  for (int j = 0; j < 8; ++j) {
    const size_t idx = (size_t)b * Ln + j * 256 + tid;
    const float a = s[j] * inv;
    a_reg[idx] = a;
    cov_out[idx] = cov_in[idx] + a;
  }
}

// ---------------- context = a_t @ h_i (two-pass, no atomics) ----------------
__global__ __launch_bounds__(256) void ctx_partial(const float* __restrict__ h_i,
                                                   const float* __restrict__ a_reg,
                                                   float* __restrict__ part) {
  __shared__ float s_a[128];
  const int blk = blockIdx.x;
  const int b = blk >> 4, ch = blk & 15, l0 = ch * 128;
  const int tid = threadIdx.x;
  if (tid < 128) s_a[tid] = a_reg[(size_t)b * Ln + l0 + tid];
  __syncthreads();
  float acc0 = 0.f, acc1 = 0.f;
  const float* hp = h_i + ((size_t)b * Ln + l0) * Dn;
  for (int i = 0; i < 128; ++i) {
    const float a = s_a[i];
    acc0 += a * hp[(size_t)i * Dn + tid];
    acc1 += a * hp[(size_t)i * Dn + tid + 256];
  }
  part[(size_t)blk * Dn + tid] = acc0;
  part[(size_t)blk * Dn + tid + 256] = acc1;
}

__global__ __launch_bounds__(256) void ctx_reduce(const float* __restrict__ part,
                                                  float* __restrict__ ctx) {
  const int b = blockIdx.x, tid = threadIdx.x;
#pragma unroll
  for (int h = 0; h < 2; ++h) {
    const int d = tid + h * 256;
    float s = 0.f;
    for (int ch = 0; ch < 16; ++ch) s += part[((size_t)(b * 16 + ch)) * Dn + d];
    ctx[(size_t)b * Dn + d] = s;
  }
}

// ---------------- launch ----------------
extern "C" void kernel_launch(void* const* d_in, const int* in_sizes, int n_in,
                              void* d_out, int out_size, void* d_ws, size_t ws_size,
                              hipStream_t stream) {
  const float* h_i  = (const float*)d_in[0];
  const float* s_t  = (const float*)d_in[1];
  const float* cov  = (const float*)d_in[2];
  const float* Wh   = (const float*)d_in[3];
  const float* bh   = (const float*)d_in[4];
  const float* Ws   = (const float*)d_in[5];
  const float* bs   = (const float*)d_in[6];
  const float* Wc   = (const float*)d_in[7];   // (1,D) -> row of 512
  const float* bc   = (const float*)d_in[8];
  const float* Vw   = (const float*)d_in[9];

  char* ws = (char*)d_ws;
  __bf16* whT = (__bf16*)ws;                    // 512*512*2   = 524288 B
  float*  v   = (float*)(ws + 524288);          // 512*4       =   2048 B
  float*  dec = (float*)(ws + 526336);          // 64*512*4    = 131072 B
  float*  prt = (float*)(ws + 657408);          // 64*16*512*4 = 2097152 B

  float* out     = (float*)d_out;
  float* ctx     = out;                 // (B,D)   32768
  float* a_reg   = out + 32768;         // (B,L)  131072 (scores -> a_t in place)
  float* cov_out = out + 163840;        // (B,L,1,1) 131072

  prep_wht  <<<Dn, 256, 0, stream>>>(Wh, whT);
  prep_v    <<<2, 256, 0, stream>>>(Vw, v);
  prep_dec  <<<Bn, 256, 0, stream>>>(s_t, Ws, bs, dec);
  attn_scores<<<(Bn * Ln) / 128, 256, 0, stream>>>(h_i, cov, whT, dec, Wc, bh, bc, v, a_reg);
  softmax_cov<<<Bn, 256, 0, stream>>>(cov, a_reg, cov_out);
  ctx_partial<<<Bn * 16, 256, 0, stream>>>(h_i, a_reg, prt);
  ctx_reduce <<<Bn, 256, 0, stream>>>(prt, ctx);
}
